// MultiheadSelfAttention_80436147519595
// MI455X (gfx1250) — matrix operand order repaired and checked by reading, not compiled, not run on hardware
//
#include <hip/hip_runtime.h>
#include <hip/hip_bf16.h>

// ---------------------------------------------------------------------------
// MI455X (gfx1250) multihead self-attention forward, bf16 WMMA pipeline
// with TDM (tensor_load_to_lds) double-buffered weight staging.
// B=2, L=2048, C=1024, H=16, HD=64, P=2.
// ---------------------------------------------------------------------------

typedef __bf16 bf16;
typedef __attribute__((ext_vector_type(16))) __bf16        v16bf;
typedef __attribute__((ext_vector_type(8)))  float         v8f;
typedef __attribute__((ext_vector_type(4)))  unsigned int  v4u;
typedef __attribute__((ext_vector_type(8)))  int           v8i;
typedef __attribute__((ext_vector_type(4)))  int           v4i;

#define B_  2
#define L_  2048
#define C_  1024
#define H_  16
#define HD_ 64
#define NQKV_ (3 * C_)
#define ML_ (B_ * L_)          // 4096 rows of x / y
#define BH_ (B_ * H_)          // 32 (b,h) pairs

#if defined(__has_builtin)
#if __has_builtin(__builtin_amdgcn_tensor_load_to_lds) && \
    __has_builtin(__builtin_amdgcn_s_wait_tensorcnt)
#define HAVE_TDM 1
#endif
#endif
#ifndef HAVE_TDM
#define HAVE_TDM 0
#endif

union FragU { v16bf v; unsigned int u[8]; };

// A-fragment (16x32 bf16, M x K), row-major source, lda in elements.
// ISA 7.12.2: lanes 0-15 -> rows 0-15 holding K {0..7,16..23};
//             lanes 16-31 -> rows 0-15 holding K {8..15,24..31}.
__device__ inline v16bf load_frag_a(const bf16* __restrict__ base, int lda, int lane) {
  FragU f;
  const int m   = lane & 15;
  const int khi = (lane >> 4) << 3;   // +8 for upper half-wave
  const unsigned int* p = reinterpret_cast<const unsigned int*>(base + m * lda);
#pragma unroll
  for (int j = 0; j < 8; ++j) {
    const int k = ((j < 4) ? 2 * j : 2 * j + 8) + khi;  // even -> dword aligned
    f.u[j] = p[k >> 1];
  }
  return f.v;
}

// B-fragment (32x16 bf16, K x N) where source holds B^T row-major:
// element (k,n) at base[n*ldb + k]. Lane n = lane&15, K {0..7,16..23} / {8..15,24..31}.
__device__ inline v16bf load_frag_b(const bf16* __restrict__ base, int ldb, int lane) {
  FragU f;
  const int n     = lane & 15;
  const int kbase = (lane >> 4) << 3; // 0 or 8
  const unsigned int* p = reinterpret_cast<const unsigned int*>(base + n * ldb + kbase);
#pragma unroll
  for (int j = 0; j < 8; ++j) f.u[j] = p[(j < 4) ? j : j + 4];
  return f.v;
}

__device__ inline v8f wmma_bf16(v16bf a, v16bf b, v8f c) {
  return __builtin_amdgcn_wmma_f32_16x16x32_bf16(
      /*neg_a=*/false, a, /*neg_b=*/false, b,
      /*c_mod=*/(short)0, c, /*reuse_a=*/false, /*reuse_b=*/false);
}

// ---------------------------------------------------------------------------
// TDM: DMA a 64(n) x 32(k) bf16 weight tile (rows stride K elements) into LDS.
// D# per ISA cdna5 ch.8: group0 = {count/lds_addr/global_addr/type},
// group1 = {data_size, tensor dims, tile dims 32x64, dim0 stride}.
// Issued by one wave; tracked with TENSORcnt.
// ---------------------------------------------------------------------------
__device__ inline void tdm_load_b_tile(const bf16* gsrc, unsigned lds_off,
                                       int K, int N) {
#if HAVE_TDM
  const unsigned long long ga = (unsigned long long)(__SIZE_TYPE__)gsrc;
  v4u g0;
  g0[0] = 1u;                                  // count=1, user descriptor
  g0[1] = lds_off;                             // lds_addr (bytes)
  g0[2] = (unsigned)(ga & 0xffffffffu);        // global_addr[31:0]
  g0[3] = (unsigned)((ga >> 32) & 0x1ffffffu)  // global_addr[56:32]
        | (2u << 30);                          // type=2 ("image")

  const unsigned long long uK = (unsigned)K, uN = (unsigned)N;
  // bits[17:16]=data_size(2B); bits[79:48]=tensor_dim0; bits[111:80]=tensor_dim1;
  // bits[127:112]=tile_dim0=32; bits[143:128]=tile_dim1=64;
  // bits[207:160]=tensor_dim0_stride=K; rest 0 (2D tile).
  const unsigned long long q0 = (1ull << 16) | (uK << 48);
  const unsigned long long q1 = (uK >> 16) | (uN << 16) | (32ull << 48);
  const unsigned long long q2 = 64ull | (uK << 32);
  const unsigned long long q3 = (uK >> 32) & 0xffffull;  // stride bits [47:32]

  v8i g1;
  g1[0] = (int)q0; g1[1] = (int)(q0 >> 32);
  g1[2] = (int)q1; g1[3] = (int)(q1 >> 32);
  g1[4] = (int)q2; g1[5] = (int)(q2 >> 32);
  g1[6] = (int)q3; g1[7] = (int)(q3 >> 32);

  const v4i zz4 = {0, 0, 0, 0};
#if __clang_major__ >= 23
  const v8i zz8 = {0, 0, 0, 0, 0, 0, 0, 0};
  __builtin_amdgcn_tensor_load_to_lds(g0, g1, zz4, zz4, zz8, 0);
#else
  __builtin_amdgcn_tensor_load_to_lds(g0, g1, zz4, zz4, 0);
#endif
#else
  (void)gsrc; (void)lds_off; (void)K; (void)N;
#endif
}

// ---------------------------------------------------------------------------
// fp32 -> bf16 elementwise conversion
// ---------------------------------------------------------------------------
__global__ __launch_bounds__(256) void cvt_bf16_kernel(
    const float* __restrict__ src, bf16* __restrict__ dst, int n) {
  int i = blockIdx.x * 256 + threadIdx.x;
  if (i < n) dst[i] = (bf16)src[i];
}

// ---------------------------------------------------------------------------
// bf16 GEMM: C[M,N] (fp32) = A[M,K] (bf16) * Bt[N,K]^T (bf16) + bias.
// Block = 8 waves covering rows [128) x cols [64). The 64x32 weight tile is
// shared by all 8 waves -> staged once per block into LDS by the TDM,
// double-buffered so the DMA overlaps the WMMAs (TENSORcnt pipelining).
// ---------------------------------------------------------------------------
__global__ __launch_bounds__(256) void gemm_bf16_kernel(
    const bf16* __restrict__ A, const bf16* __restrict__ Bt,
    const float* __restrict__ bias, float* __restrict__ C,
    int M, int N, int K) {
  __shared__ bf16 sB[2][64 * 32];   // 2 x 4KB weight tiles

  const int tid  = threadIdx.x;
  const int lane = tid & 31;
  const int wid  = tid >> 5;
  const int m0   = blockIdx.y * 128 + wid * 16;
  const int n0   = blockIdx.x * 64;

  const v8f z = {0.f, 0.f, 0.f, 0.f, 0.f, 0.f, 0.f, 0.f};
  v8f acc[4] = {z, z, z, z};

#if HAVE_TDM
  if (wid == 0)
    tdm_load_b_tile(Bt + (long)n0 * K, (unsigned)(__SIZE_TYPE__)&sB[0][0], K, N);
#else
  {  // cooperative fallback: 256 threads x 8 bf16 = 4KB
    const int elt = tid * 8, n = elt >> 5, k = elt & 31;
    const unsigned int* s = (const unsigned int*)(Bt + (long)(n0 + n) * K + k);
    unsigned int* d = (unsigned int*)(&sB[0][elt]);
    d[0] = s[0]; d[1] = s[1]; d[2] = s[2]; d[3] = s[3];
  }
#endif

  for (int k0 = 0; k0 < K; k0 += 32) {
    const int cur = (k0 >> 5) & 1;
    const bool more = (k0 + 32) < K;
#if HAVE_TDM
    if (wid == 0) {
      if (more) {
        tdm_load_b_tile(Bt + (long)n0 * K + (k0 + 32),
                        (unsigned)(__SIZE_TYPE__)&sB[cur ^ 1][0], K, N);
        __builtin_amdgcn_s_wait_tensorcnt((short)1);  // tile 'cur' landed
      } else {
        __builtin_amdgcn_s_wait_tensorcnt((short)0);
      }
    }
#else
    if (more) {
      const int elt = tid * 8, n = elt >> 5, k = elt & 31;
      const unsigned int* s =
          (const unsigned int*)(Bt + (long)(n0 + n) * K + (k0 + 32) + k);
      unsigned int* d = (unsigned int*)(&sB[cur ^ 1][elt]);
      d[0] = s[0]; d[1] = s[1]; d[2] = s[2]; d[3] = s[3];
    }
#endif
    __syncthreads();   // tile[cur] visible to all waves

    if (more) __builtin_prefetch(A + (long)m0 * K + k0 + 32, 0, 3);
    const v16bf a = load_frag_a(A + (long)m0 * K + k0, K, lane);
    const bf16* sb = &sB[cur][0];
#pragma unroll
    for (int nt = 0; nt < 4; ++nt) {
      const v16bf b = load_frag_b(sb + nt * 16 * 32, 32, lane);
      acc[nt] = wmma_bf16(a, b, acc[nt]);
    }
    __syncthreads();   // protect tile[cur] from next DMA overwrite
  }

#pragma unroll
  for (int nt = 0; nt < 4; ++nt) {
    const int col = n0 + nt * 16 + (lane & 15);
    const float bv = bias ? bias[col] : 0.0f;
#pragma unroll
    for (int r = 0; r < 8; ++r) {
      const int row = m0 + r + ((lane >> 4) << 3);
      C[(long)row * N + col] = acc[nt][r] + bv;
    }
  }
}

// ---------------------------------------------------------------------------
// Fused RMSNorm + RoPE + layout shuffle.
// One wave per (b,l,h) row; lane owns RoPE pair (2*lane, 2*lane+1).
// Writes q,k as bf16 [BH, L, HD]; v transposed as bf16 [BH, HD, L].
// ---------------------------------------------------------------------------
__global__ __launch_bounds__(256) void qkv_post_kernel(
    const float* __restrict__ qkv, const float* __restrict__ pos,
    const float* __restrict__ Wtheta,
    bf16* __restrict__ qb, bf16* __restrict__ kbuf, bf16* __restrict__ vt) {
  const int lane = threadIdx.x & 31;
  const int w    = blockIdx.x * 8 + (threadIdx.x >> 5);   // 0 .. B*L*H-1
  const int h    = w & (H_ - 1);
  const int l    = (w >> 4) & (L_ - 1);
  const int b    = w >> 15;
  const int bh   = b * H_ + h;

  const long base = (long)(b * L_ + l) * NQKV_ + h * HD_ + 2 * lane;
  float qre = qkv[base +      0], qim = qkv[base +      1];
  float kre = qkv[base +   C_ ], kim = qkv[base +   C_ + 1];
  float vre = qkv[base + 2*C_ ], vim = qkv[base + 2*C_ + 1];

  float sq = qre * qre + qim * qim;
  float sk = kre * kre + kim * kim;
#pragma unroll
  for (int off = 16; off > 0; off >>= 1) {
    sq += __shfl_xor(sq, off, 32);
    sk += __shfl_xor(sk, off, 32);
  }
  const float qinv = rsqrtf(sq * (1.0f / HD_) + 1e-5f);
  const float kinv = rsqrtf(sk * (1.0f / HD_) + 1e-5f);
  qre *= qinv; qim *= qinv; kre *= kinv; kim *= kinv;

  const int j = h * (HD_ / 2) + lane;
  const float p0 = pos[(long)(b * L_ + l) * 2 + 0];
  const float p1 = pos[(long)(b * L_ + l) * 2 + 1];
  const float th = p0 * Wtheta[2 * j + 0] + p1 * Wtheta[2 * j + 1];
  const float c = __cosf(th), s = __sinf(th);

  const long qdst = ((long)bh * L_ + l) * HD_ + 2 * lane;
  qb[qdst]     = (bf16)(qre * c - qim * s);
  qb[qdst + 1] = (bf16)(qre * s + qim * c);
  kbuf[qdst]     = (bf16)(kre * c - kim * s);
  kbuf[qdst + 1] = (bf16)(kre * s + kim * c);

  const long vdst = ((long)bh * HD_ + 2 * lane) * L_ + l;
  vt[vdst]      = (bf16)vre;
  vt[vdst + L_] = (bf16)vim;
}

// ---------------------------------------------------------------------------
// Flash attention: one wave per (bh, 16-query tile). Online softmax.
// Per 32-key step: 4 WMMA for S = Q K^T, 4 WMMA for O += P V.
// P converted D-layout -> A-layout through per-wave LDS tile (1 KB).
// ---------------------------------------------------------------------------
__global__ __launch_bounds__(256) void attn_kernel(
    const bf16* __restrict__ qb, const bf16* __restrict__ kbuf,
    const bf16* __restrict__ vt, bf16* __restrict__ yb) {
  __shared__ bf16 sP[8 * 16 * 32];

  const int lane = threadIdx.x & 31;
  const int wid  = threadIdx.x >> 5;
  const int w    = blockIdx.x * 8 + wid;      // 0 .. BH*(L/16)-1
  const int qt   = w & (L_ / 16 - 1);         // query tile
  const int bh   = w >> 7;
  const int q0   = qt * 16;
  const int b    = bh >> 4, h = bh & 15;
  bf16* myP = sP + wid * (16 * 32);

  const bf16* Qbase = qb + ((long)bh * L_ + q0) * HD_;
  const v16bf qa0 = load_frag_a(Qbase +  0, HD_, lane);
  const v16bf qa1 = load_frag_a(Qbase + 32, HD_, lane);

  const v8f z = {0.f, 0.f, 0.f, 0.f, 0.f, 0.f, 0.f, 0.f};
  v8f o[4] = {z, z, z, z};
  float mrun[8], lrun[8];
#pragma unroll
  for (int r = 0; r < 8; ++r) { mrun[r] = -1e30f; lrun[r] = 0.f; }

  const float scale = 0.125f;  // 1/sqrt(64)
  const int rowhalf = (lane >> 4) << 3;   // 0 or 8
  const int colid   = lane & 15;

  for (int k0 = 0; k0 < L_; k0 += 32) {
    const bf16* K0 = kbuf + ((long)bh * L_ + k0) * HD_;
    v8f s0 = z, s1 = z;
    s0 = wmma_bf16(qa0, load_frag_b(K0 +  0,            HD_, lane), s0);
    s0 = wmma_bf16(qa1, load_frag_b(K0 + 32,            HD_, lane), s0);
    s1 = wmma_bf16(qa0, load_frag_b(K0 + 16 * HD_ +  0, HD_, lane), s1);
    s1 = wmma_bf16(qa1, load_frag_b(K0 + 16 * HD_ + 32, HD_, lane), s1);

    // online softmax update; row r+rowhalf stats replicated in each 16-lane half
#pragma unroll
    for (int r = 0; r < 8; ++r) {
      const float v0 = s0[r] * scale, v1 = s1[r] * scale;
      float t = fmaxf(v0, v1);
#pragma unroll
      for (int off = 8; off >= 1; off >>= 1) t = fmaxf(t, __shfl_xor(t, off, 32));
      const float mnew  = fmaxf(mrun[r], t);
      const float alpha = __expf(mrun[r] - mnew);
      const float p0 = __expf(v0 - mnew), p1 = __expf(v1 - mnew);
      float rs = p0 + p1;
#pragma unroll
      for (int off = 8; off >= 1; off >>= 1) rs += __shfl_xor(rs, off, 32);
      lrun[r] = lrun[r] * alpha + rs;
      mrun[r] = mnew;
      o[0][r] *= alpha; o[1][r] *= alpha; o[2][r] *= alpha; o[3][r] *= alpha;
      const int m = r + rowhalf;
      myP[m * 32 + colid]      = (bf16)p0;
      myP[m * 32 + colid + 16] = (bf16)p1;
    }
    __syncthreads();   // D-layout -> A-layout crossing through LDS

    const v16bf pa = load_frag_a(myP, 32, lane);
#pragma unroll
    for (int t = 0; t < 4; ++t) {
      const bf16* Vb = vt + ((long)bh * HD_ + t * 16) * L_ + k0;
      o[t] = wmma_bf16(pa, load_frag_b(Vb, L_, lane), o[t]);
    }
    __syncthreads();
  }

  // normalize + store y[b, l, h*HD + hd] as bf16 (A-operand of output GEMM)
#pragma unroll
  for (int t = 0; t < 4; ++t) {
#pragma unroll
    for (int r = 0; r < 8; ++r) {
      const int row = q0 + r + rowhalf;
      const int col = h * HD_ + t * 16 + colid;
      yb[(long)(b * L_ + row) * C_ + col] = (bf16)(o[t][r] / lrun[r]);
    }
  }
}

// ---------------------------------------------------------------------------
// Host-side orchestration
// ---------------------------------------------------------------------------
extern "C" void kernel_launch(void* const* d_in, const int* in_sizes, int n_in,
                              void* d_out, int out_size, void* d_ws, size_t ws_size,
                              hipStream_t stream) {
  const float* x      = (const float*)d_in[0];  // [B,L,C]
  const float* pos    = (const float*)d_in[1];  // [B,L,2]
  const float* Wqkv   = (const float*)d_in[2];  // [3C,C]
  const float* bqkv   = (const float*)d_in[3];  // [3C]
  const float* Wtheta = (const float*)d_in[4];  // [C/2,2]
  const float* Wy     = (const float*)d_in[5];  // [C,C]
  float* out = (float*)d_out;                   // [B,L,C]

  const long nX    = (long)ML_ * C_;        // 4194304
  const long nWqkv = (long)NQKV_ * C_;      // 3145728
  const long nWy   = (long)C_ * C_;         // 1048576
  const long nQKV  = (long)ML_ * NQKV_;     // 12582912
  const long nHead = (long)BH_ * L_ * HD_;  // 4194304

  char* p = (char*)d_ws;
  bf16*  xb     = (bf16*)p;  p += nX * sizeof(bf16);
  bf16*  wqkvb  = (bf16*)p;  p += nWqkv * sizeof(bf16);
  bf16*  wyb    = (bf16*)p;  p += nWy * sizeof(bf16);
  float* qkvf   = (float*)p; p += nQKV * sizeof(float);
  bf16*  qbuf   = (bf16*)p;  p += nHead * sizeof(bf16);
  bf16*  kbuf   = (bf16*)p;  p += nHead * sizeof(bf16);
  bf16*  vtbuf  = (bf16*)p;  p += nHead * sizeof(bf16);
  bf16*  ybuf   = (bf16*)p;  p += nHead * sizeof(bf16);
  (void)ws_size; (void)in_sizes; (void)n_in; (void)out_size;

  // 1) downconvert operands to bf16
  cvt_bf16_kernel<<<(int)(nX    / 256), 256, 0, stream>>>(x,    xb,    (int)nX);
  cvt_bf16_kernel<<<(int)(nWqkv / 256), 256, 0, stream>>>(Wqkv, wqkvb, (int)nWqkv);
  cvt_bf16_kernel<<<(int)(nWy   / 256), 256, 0, stream>>>(Wy,   wyb,   (int)nWy);

  // 2) QKV projection: [4096,1024] x [3072,1024]^T -> fp32 [4096,3072]
  {
    dim3 grid(NQKV_ / 64, ML_ / 128);
    gemm_bf16_kernel<<<grid, 256, 0, stream>>>(xb, wqkvb, bqkv, qkvf,
                                               ML_, NQKV_, C_);
  }

  // 3) RMSNorm + RoPE + head-major relayout (V transposed)
  qkv_post_kernel<<<(B_ * L_ * H_) / 8, 256, 0, stream>>>(
      qkvf, pos, Wtheta, qbuf, kbuf, vtbuf);

  // 4) flash attention: BH * (L/16) waves, 8 waves/block
  attn_kernel<<<(BH_ * (L_ / 16)) / 8, 256, 0, stream>>>(qbuf, kbuf, vtbuf, ybuf);

  // 5) output projection: [4096,1024] x [1024,1024]^T -> fp32 d_out
  {
    dim3 grid(C_ / 64, ML_ / 128);
    gemm_bf16_kernel<<<grid, 256, 0, stream>>>(ybuf, wyb, nullptr, out,
                                               ML_, C_, C_);
  }
}
